// QwenNextAttention_16862041604494
// MI455X (gfx1250) — matrix operand order repaired and checked by reading, not compiled, not run on hardware
//
#include <hip/hip_runtime.h>
#include <hip/hip_bf16.h>
#include <math.h>

// ---------------------------------------------------------------------------
// QwenNext attention block for MI455X (gfx1250, wave32, WMMA 16x16x32 f16,
// Tensor Data Mover staging into LDS, double-buffered via TENSORcnt)
//   B=2, N=2048, HIDDEN=896, H_Q=14, H_KV=2, D=64, rope base 1e6
// ---------------------------------------------------------------------------

typedef __attribute__((ext_vector_type(16))) _Float16     v16h_t;
typedef __attribute__((ext_vector_type(8)))  float        v8f_t;
typedef __attribute__((ext_vector_type(4)))  unsigned int v4u_t;
typedef __attribute__((ext_vector_type(8)))  unsigned int v8u_t;

union V16H { v16h_t v; _Float16 e[16]; };
union V8F  { v8f_t  v; float    e[8];  };

#define WMMA_F16(A, B, C) \
  __builtin_amdgcn_wmma_f32_16x16x32_f16(false, (A), false, (B), (short)0, (C), false, false)

constexpr int Bb  = 2;
constexpr int Nn  = 2048;
constexpr int HID = 896;
constexpr int HQ  = 14;
constexpr int HKV = 2;
constexpr int Dd  = 64;
constexpr int M_TOT = Bb * Nn;          // 4096 rows
constexpr int QDIM  = HQ * Dd;          // 896
constexpr int KDIM  = HKV * Dd;         // 128

// ---------------------------------------------------------------------------
// LDS byte offset of a __shared__ object: in the LDS aperture the low 32 bits
// of the flat address are the LDS offset (ISA 10.2 aperture mapping).
// ---------------------------------------------------------------------------
__device__ __forceinline__ unsigned lds_off(const void* p) {
  return (unsigned)(unsigned long long)p;
}

// ---------------------------------------------------------------------------
// TENSOR_LOAD_TO_LDS with a 2D descriptor (groups 2/3 NULL).
// D# group0: [count=1 | lds_addr | global_addr(57b) | type=2]
// D# group1: [wg_mask=0, data_size=2B | tensor_dim0/1 | tile_dim0/1/2 |
//             tensor_dim0_stride | tensor_dim1_stride=0]
// dims/strides in data_size (2-byte) units. All dwords forced to SGPRs.
// ---------------------------------------------------------------------------
__device__ __forceinline__ void tdm_load_2d_f16(
    unsigned lds_byte_off, const _Float16* gaddr,
    unsigned tensor_d0, unsigned tensor_d1, unsigned stride0,
    unsigned tile_d0, unsigned tile_d1)
{
  const unsigned long long ga = (unsigned long long)gaddr;
  const unsigned g0w0 = __builtin_amdgcn_readfirstlane(1u);               // count=1
  const unsigned g0w1 = __builtin_amdgcn_readfirstlane(lds_byte_off);
  const unsigned g0w2 = __builtin_amdgcn_readfirstlane((unsigned)ga);
  const unsigned g0w3 = __builtin_amdgcn_readfirstlane(
      ((unsigned)(ga >> 32) & 0x01FFFFFFu) | (2u << 30));                 // type=2
  const unsigned g1w0 = __builtin_amdgcn_readfirstlane(1u << 16);         // 2B elems
  const unsigned g1w1 = __builtin_amdgcn_readfirstlane(tensor_d0 << 16);
  const unsigned g1w2 = __builtin_amdgcn_readfirstlane((tensor_d0 >> 16) | (tensor_d1 << 16));
  const unsigned g1w3 = __builtin_amdgcn_readfirstlane((tensor_d1 >> 16) | (tile_d0 << 16));
  const unsigned g1w4 = __builtin_amdgcn_readfirstlane(tile_d1 & 0xFFFFu);// tile_dim2=0
  const unsigned g1w5 = __builtin_amdgcn_readfirstlane(stride0);
  const unsigned g1w6 = __builtin_amdgcn_readfirstlane(0u);
  const unsigned g1w7 = __builtin_amdgcn_readfirstlane(0u);
  const v4u_t g0 = { g0w0, g0w1, g0w2, g0w3 };
  const v8u_t g1 = { g1w0, g1w1, g1w2, g1w3, g1w4, g1w5, g1w6, g1w7 };
  asm volatile("tensor_load_to_lds %0, %1" :: "s"(g0), "s"(g1) : "memory");
}

// ---------------------------------------------------------------------------
// fp32 -> fp16 conversion (grid-stride)
// ---------------------------------------------------------------------------
__global__ void cvt_f32_f16(const float* __restrict__ in, _Float16* __restrict__ out, int n) {
  for (int i = blockIdx.x * blockDim.x + threadIdx.x; i < n; i += gridDim.x * blockDim.x)
    out[i] = (_Float16)in[i];
}

// ---------------------------------------------------------------------------
// WMMA GEMM:  C[M x Nout] = A[M x K] * W[K x Nout] (+bias)
// Block = 256 threads = 8 waves -> 128x64 C tile (wave w: rows w*16..w*16+15,
// all 64 cols, 4 accumulators). Per 32-wide K-step, wave0 TDMs the A tile
// (128x32, row stride K) and wave1 TDMs the B tile (32x64, row stride Nout)
// into double-buffered LDS; s_wait_tensorcnt pipelines copy against compute.
// WMMA register layouts (CDNA5):
//   A 16x32 f16 : lane L row M=L%16, elem e -> k = (L/16)*16 + e (this code's order, as B)
//   B 32x16 f16 : lane L col N=L%16, elem e -> k = (L/16)*16 + e (this code's order, same as A; not AMD's table order)
//   C 16x16 f32 : vgpr r, lane L   -> row r + 8*(L/16), col L%16
// ---------------------------------------------------------------------------
template <bool OUT_F16>
__global__ void wmma_gemm(const _Float16* __restrict__ A, const _Float16* __restrict__ W,
                          const float* __restrict__ bias,
                          float* __restrict__ Cf, _Float16* __restrict__ Ch,
                          int M, int K, int Nout) {
  __shared__ _Float16 At[2][128][32];
  __shared__ _Float16 Bt[2][32][64];

  const int lane = threadIdx.x & 31;
  const int wave = threadIdx.x >> 5;
  const int half = lane >> 4;
  const int l15  = lane & 15;
  const int n0   = blockIdx.x * 64;
  const int m0   = blockIdx.y * 128;

  V8F acc[4];
#pragma unroll
  for (int c = 0; c < 4; ++c)
#pragma unroll
    for (int r = 0; r < 8; ++r) acc[c].e[r] = 0.0f;

  const int NS = K / 32;
  // prologue: stage K-step 0
  if (wave == 0)
    tdm_load_2d_f16(lds_off(&At[0][0][0]), A + (size_t)m0 * K, 32, 128, (unsigned)K, 32, 128);
  if (wave == 1)
    tdm_load_2d_f16(lds_off(&Bt[0][0][0]), W + (size_t)0 * Nout + n0, 64, 32, (unsigned)Nout, 64, 32);

  for (int s = 0; s < NS; ++s) {
    if (s > 0) __syncthreads();               // readers of buf (s+1)&1 are done
    if (s + 1 < NS) {
      const int k1 = (s + 1) * 32;
      if (wave == 0)
        tdm_load_2d_f16(lds_off(&At[(s + 1) & 1][0][0]),
                        A + (size_t)m0 * K + k1, 32, 128, (unsigned)K, 32, 128);
      if (wave == 1)
        tdm_load_2d_f16(lds_off(&Bt[(s + 1) & 1][0][0]),
                        W + (size_t)k1 * Nout + n0, 64, 32, (unsigned)Nout, 64, 32);
      if (wave < 2) __builtin_amdgcn_s_wait_tensorcnt((short)1);   // step s landed
    } else {
      if (wave < 2) __builtin_amdgcn_s_wait_tensorcnt((short)0);
    }
    __syncthreads();                          // publish LDS tiles to all waves

    const int bf = s & 1;
    V16H a;
#pragma unroll
    for (int e = 0; e < 16; ++e)
      a.e[e] = At[bf][wave * 16 + l15][half * 16 + e];
#pragma unroll
    for (int cc = 0; cc < 4; ++cc) {
      V16H b;
#pragma unroll
      for (int e = 0; e < 16; ++e)
        b.e[e] = Bt[bf][half * 16 + e][cc * 16 + l15];
      acc[cc].v = WMMA_F16(a.v, b.v, acc[cc].v);
    }
  }

#pragma unroll
  for (int cc = 0; cc < 4; ++cc) {
    const int col = n0 + cc * 16 + l15;
    const float bv = bias ? bias[col] : 0.0f;
#pragma unroll
    for (int r = 0; r < 8; ++r) {
      const int row = m0 + wave * 16 + r + 8 * half;
      const float v = acc[cc].e[r] + bv;
      if (OUT_F16) Ch[(size_t)row * Nout + col] = (_Float16)v;
      else         Cf[(size_t)row * Nout + col] = v;
    }
  }
}

// ---------------------------------------------------------------------------
// RoPE (base 1e6) in place on f16 Q [B,N,HQ,D] and K [B,N,HKV,D].
// ---------------------------------------------------------------------------
__global__ void rope_kernel(_Float16* __restrict__ Q, _Float16* __restrict__ Kc,
                            const int* __restrict__ pos_ids) {
  const int t = blockIdx.x * blockDim.x + threadIdx.x;   // [0, B*N*16*32)
  const int i = t & 31;
  const int h = (t >> 5) & 15;
  const int n = (t >> 9) & (Nn - 1);
  const int b = t >> 20;

  const float pos = (float)pos_ids[b * Nn + n];
  const float inv = __expf(-0.431734705f * (float)i);    // 1e6^(-i/32)
  const float f   = pos * inv;
  float sn, cs;
  __sincosf(f, &sn, &cs);

  _Float16* p;
  if (h < HQ) p = Q  + ((size_t)(b * Nn + n) * HQ  + h)        * Dd;
  else        p = Kc + ((size_t)(b * Nn + n) * HKV + (h - HQ)) * Dd;

  const float x1 = (float)p[i];
  const float x2 = (float)p[i + 32];
  p[i]      = (_Float16)(x1 * cs - x2 * sn);
  p[i + 32] = (_Float16)(x2 * cs + x1 * sn);
}

// ---------------------------------------------------------------------------
// Causal GQA flash attention. One wave owns a 16-query x D=64 tile of a head.
// Per 32-key block: K/V tiles (32x64 f16) TDM-loaded into per-wave LDS double
// buffers (issue-ahead + s_wait_tensorcnt); 4 WMMAs for S=QK^T, online
// softmax with 16-lane shfl reductions, P relayout via LDS, 4 WMMAs for PV.
// No cross-wave barriers: every wave uses only its own LDS slices.
// ---------------------------------------------------------------------------
__global__ void flash_attn(const _Float16* __restrict__ Q, const _Float16* __restrict__ Kc,
                           const _Float16* __restrict__ Vc, _Float16* __restrict__ Oa) {
  __shared__ _Float16 kvs[4][2][2][32 * 64];  // [wave][buf][K/V][tile]
  __shared__ _Float16 pTile[4][16][32];       // per-wave P bounce (C->A layout)

  const int lane = threadIdx.x & 31;
  const int wave = threadIdx.x >> 5;
  const int half = lane >> 4;
  const int l15  = lane & 15;

  const int tile = blockIdx.x * 4 + wave;     // 0 .. B*HQ*(N/16)-1
  const int qt   = tile & 127;                // N/16 = 128
  const int h    = (tile >> 7) % HQ;
  const int b    = tile / (128 * HQ);
  const int hk   = h / (HQ / HKV);
  const int q0   = qt * 16;

  // Q tile in A layout, split by contraction half (d 0..31 / 32..63)
  V16H a0, a1;
  const _Float16* qbase = Q + ((size_t)(b * Nn + q0 + l15) * HQ + h) * Dd;
#pragma unroll
  for (int e = 0; e < 16; ++e) {
    const int d = half * 16 + e;
    a0.e[e] = qbase[d];
    a1.e[e] = qbase[d + 32];
  }

  float m[8], l[8];
  V8F acc[4];
#pragma unroll
  for (int r = 0; r < 8; ++r) { m[r] = -INFINITY; l[r] = 0.0f; }
#pragma unroll
  for (int c = 0; c < 4; ++c)
#pragma unroll
    for (int r = 0; r < 8; ++r) acc[c].e[r] = 0.0f;

  V8F zf;
#pragma unroll
  for (int r = 0; r < 8; ++r) zf.e[r] = 0.0f;

  auto issueKV = [&](int ikb) {
    const int kb = ikb * 32;
    const _Float16* kg = Kc + ((size_t)(b * Nn + kb) * HKV + hk) * Dd;
    const _Float16* vg = Vc + ((size_t)(b * Nn + kb) * HKV + hk) * Dd;
    tdm_load_2d_f16(lds_off(&kvs[wave][ikb & 1][0][0]), kg, 64, 32, KDIM, 64, 32);
    tdm_load_2d_f16(lds_off(&kvs[wave][ikb & 1][1][0]), vg, 64, 32, KDIM, 64, 32);
  };

  const int nkb = (q0 + 16 + 31) / 32;        // key blocks needed (causal)
  issueKV(0);

  for (int ikb = 0; ikb < nkb; ++ikb) {
    const int kb = ikb * 32;
    if (ikb + 1 < nkb) {
      issueKV(ikb + 1);
      __builtin_amdgcn_s_wait_tensorcnt((short)2);   // tile ikb landed
    } else {
      __builtin_amdgcn_s_wait_tensorcnt((short)0);
    }
    asm volatile("" ::: "memory");

    const _Float16* kt = &kvs[wave][ikb & 1][0][0];  // [key][d] 32x64
    const _Float16* vt = &kvs[wave][ikb & 1][1][0];

    // ---- S = Q K^T (two 16-key tiles), contraction over d=64 -------------
    V16H bk;
    V8F s0, s1;
#pragma unroll
    for (int e = 0; e < 16; ++e) bk.e[e] = kt[l15 * 64 + half * 16 + e];
    s0.v = WMMA_F16(a0.v, bk.v, zf.v);
#pragma unroll
    for (int e = 0; e < 16; ++e) bk.e[e] = kt[l15 * 64 + 32 + half * 16 + e];
    s0.v = WMMA_F16(a1.v, bk.v, s0.v);
#pragma unroll
    for (int e = 0; e < 16; ++e) bk.e[e] = kt[(16 + l15) * 64 + half * 16 + e];
    s1.v = WMMA_F16(a0.v, bk.v, zf.v);
#pragma unroll
    for (int e = 0; e < 16; ++e) bk.e[e] = kt[(16 + l15) * 64 + 32 + half * 16 + e];
    s1.v = WMMA_F16(a1.v, bk.v, s1.v);

    // ---- online softmax (row r + 8*half lives across a 16-lane half) -----
#pragma unroll
    for (int r = 0; r < 8; ++r) {
      const int qg = q0 + r + 8 * half;
      float x0 = s0.e[r] * 0.125f;                   // 1/sqrt(64)
      float x1 = s1.e[r] * 0.125f;
      if (kb + l15 > qg)      x0 = -1e30f;
      if (kb + 16 + l15 > qg) x1 = -1e30f;

      float rm = fmaxf(x0, x1);
      rm = fmaxf(rm, __shfl_xor(rm, 1));
      rm = fmaxf(rm, __shfl_xor(rm, 2));
      rm = fmaxf(rm, __shfl_xor(rm, 4));
      rm = fmaxf(rm, __shfl_xor(rm, 8));
      const float mn = fmaxf(m[r], rm);

      const float p0 = __expf(x0 - mn);
      const float p1 = __expf(x1 - mn);
      float rs = p0 + p1;
      rs += __shfl_xor(rs, 1);
      rs += __shfl_xor(rs, 2);
      rs += __shfl_xor(rs, 4);
      rs += __shfl_xor(rs, 8);

      const float corr = __expf(m[r] - mn);          // exp(-inf)=0 first block
      l[r] = l[r] * corr + rs;
      m[r] = mn;
#pragma unroll
      for (int c = 0; c < 4; ++c) acc[c].e[r] *= corr;

      pTile[wave][r + 8 * half][l15]      = (_Float16)p0;
      pTile[wave][r + 8 * half][16 + l15] = (_Float16)p1;
    }

    asm volatile("s_wait_dscnt 0" ::: "memory");     // wave-private LDS bounce
    V16H ap;
#pragma unroll
    for (int e = 0; e < 16; ++e)
      ap.e[e] = pTile[wave][l15][half * 16 + e];

    // ---- O += P V (4 column chunks of D=64) ------------------------------
#pragma unroll
    for (int c = 0; c < 4; ++c) {
      V16H bv;
#pragma unroll
      for (int e = 0; e < 16; ++e)
        bv.e[e] = vt[(half * 16 + e) * 64 + c * 16 + l15];
      acc[c].v = WMMA_F16(ap.v, bv.v, acc[c].v);
    }
  }

  // ---- epilogue: O / (l + 1e-8), store f16 [B*N, HQ*D] -------------------
#pragma unroll
  for (int r = 0; r < 8; ++r) {
    const float inv_l = 1.0f / (l[r] + 1e-8f);
    const int row = b * Nn + q0 + r + 8 * half;
#pragma unroll
    for (int c = 0; c < 4; ++c)
      Oa[(size_t)row * QDIM + h * Dd + c * 16 + l15] = (_Float16)(acc[c].e[r] * inv_l);
  }
}

// ---------------------------------------------------------------------------
// Host-side orchestration
// ---------------------------------------------------------------------------
extern "C" void kernel_launch(void* const* d_in, const int* in_sizes, int n_in,
                              void* d_out, int out_size, void* d_ws, size_t ws_size,
                              hipStream_t stream) {
  const float* hidden = (const float*)d_in[0];
  const int*   posids = (const int*)  d_in[1];
  const float* q_w    = (const float*)d_in[2];
  const float* q_b    = (const float*)d_in[3];
  const float* k_w    = (const float*)d_in[4];
  const float* k_b    = (const float*)d_in[5];
  const float* v_w    = (const float*)d_in[6];
  const float* v_b    = (const float*)d_in[7];
  const float* o_w    = (const float*)d_in[8];
  float* out = (float*)d_out;

  char* ws = (char*)d_ws;
  size_t off = 0;
  auto carve = [&](size_t bytes) -> _Float16* {
    _Float16* p = (_Float16*)(ws + off);
    off += (bytes + 255) & ~(size_t)255;
    return p;
  };
  _Float16* hid16 = carve((size_t)M_TOT * HID  * 2);
  _Float16* qw16  = carve((size_t)HID   * QDIM * 2);
  _Float16* kw16  = carve((size_t)HID   * KDIM * 2);
  _Float16* vw16  = carve((size_t)HID   * KDIM * 2);
  _Float16* ow16  = carve((size_t)QDIM  * HID  * 2);
  _Float16* Q16   = carve((size_t)M_TOT * QDIM * 2);
  _Float16* K16   = carve((size_t)M_TOT * KDIM * 2);
  _Float16* V16   = carve((size_t)M_TOT * KDIM * 2);
  _Float16* AO16  = carve((size_t)M_TOT * QDIM * 2);

  auto cvt = [&](const float* src, _Float16* dst, int n) {
    cvt_f32_f16<<<(n + 255) / 256, 256, 0, stream>>>(src, dst, n);
  };
  cvt(hidden, hid16, M_TOT * HID);
  cvt(q_w, qw16, HID * QDIM);
  cvt(k_w, kw16, HID * KDIM);
  cvt(v_w, vw16, HID * KDIM);
  cvt(o_w, ow16, QDIM * HID);

  // QKV projections (f16 out, bias added): 128x64 tile per 256-thread block
  wmma_gemm<true><<<dim3(QDIM / 64, M_TOT / 128), 256, 0, stream>>>(
      hid16, qw16, q_b, nullptr, Q16, M_TOT, HID, QDIM);
  wmma_gemm<true><<<dim3(KDIM / 64, M_TOT / 128), 256, 0, stream>>>(
      hid16, kw16, k_b, nullptr, K16, M_TOT, HID, KDIM);
  wmma_gemm<true><<<dim3(KDIM / 64, M_TOT / 128), 256, 0, stream>>>(
      hid16, vw16, v_b, nullptr, V16, M_TOT, HID, KDIM);

  rope_kernel<<<(Bb * Nn * 16 * 32) / 256, 256, 0, stream>>>(Q16, K16, posids);

  flash_attn<<<(Bb * HQ * (Nn / 16)) / 4, 128, 0, stream>>>(Q16, K16, V16, AO16);

  wmma_gemm<false><<<dim3(HID / 64, M_TOT / 128), 256, 0, stream>>>(
      AO16, ow16, nullptr, out, nullptr, M_TOT, QDIM, HID);
}
